// CCALayer_51307679318761
// MI455X (gfx1250) — compile-verified, matrix operand order checked by reading, not run on hardware
//
#include <hip/hip_runtime.h>
#include <hip/hip_bf16.h>

typedef _Float16 half_t;
typedef __attribute__((ext_vector_type(16))) _Float16 v16h;
typedef __attribute__((ext_vector_type(8)))  _Float16 v8h;
typedef __attribute__((ext_vector_type(8)))  float    v8f;

// -------- helpers --------
__device__ __forceinline__ v16h frag_from(const half_t* lo, const half_t* hi) {
  v8h a = *(const v8h*)lo;
  v8h b = *(const v8h*)hi;
  v16h r;
#pragma unroll
  for (int i = 0; i < 8; ++i) { r[i] = a[i]; r[8 + i] = b[i]; }
  return r;
}

__device__ __forceinline__ v8f wmma_f16(v16h a, v16h b, v8f c) {
  return __builtin_amdgcn_wmma_f32_16x16x32_f16(false, a, false, b, (short)0, c, false, false);
}

// -------- weight packing: w_1x1 -> f16, c2 convs -> Wp[co][tap*16+ci] (K padded 144->160)
__global__ void k_pack(const float* __restrict__ w1x1, const float* __restrict__ w2c2,
                       const float* __restrict__ w1c2, half_t* __restrict__ wh,
                       half_t* __restrict__ Wp2, half_t* __restrict__ Wp1) {
  int idx = blockIdx.x * blockDim.x + threadIdx.x;
  if (idx < 64 * 640) { wh[idx] = (half_t)w1x1[idx]; return; }
  idx -= 64 * 640;
  if (idx < 16 * 160) {
    int co = idx / 160, k = idx % 160, tap = k / 16, ci = k % 16;
    Wp2[idx] = (half_t)((tap < 9) ? w2c2[co * 144 + ci * 9 + tap] : 0.f);
    return;
  }
  idx -= 16 * 160;
  if (idx < 16 * 160) {
    int co = idx / 160, k = idx % 160, tap = k / 16, ci = k % 16;
    Wp1[idx] = (half_t)((tap < 9) ? w1c2[co * 144 + ci * 9 + tap] : 0.f);
  }
}

// -------- stage A: channel-center, convert to f16 pixel-major [img][112][640]
__global__ void k_center(const float* __restrict__ spt, const float* __restrict__ qry,
                         half_t* __restrict__ xh, float* __restrict__ meanb) {
  int idx = blockIdx.x * blockDim.x + threadIdx.x;
  if (idx >= 8000) return;
  int img = idx / 100, p = idx % 100;
  const float* src = (img < 5) ? (spt + (size_t)img * 640 * 100)
                               : (qry + (size_t)(img - 5) * 640 * 100);
  float s = 0.f;
  for (int c = 0; c < 640; ++c) s += src[c * 100 + p];
  float mn = s * (1.f / 640.f);
  meanb[idx] = mn;
  half_t* dst = xh + ((size_t)img * 112 + p) * 640;
  for (int c = 0; c < 640; ++c) dst[c] = (half_t)(src[c * 100 + p] - mn);
}

// -------- stage A: 1x1 projection GEMM via WMMA (4 M-tiles/wave, double-buffered K),
// BN + relu, -> feat[img][112][64].  Launch: <<<80, 224>>> (7 waves = 7 N-tiles)
__global__ void k_gemm_feat(const half_t* __restrict__ wh, const half_t* __restrict__ xh,
                            const float* __restrict__ bn, float* __restrict__ feat) {
  int img = blockIdx.x;
  int tid = threadIdx.x, wave = tid >> 5, lane = tid & 31;
  int m = lane & 15, g = lane >> 4;
  int nt = wave;                       // 0..6
  int col = nt * 16 + m;
  const half_t* brow = xh + ((size_t)img * 112 + col) * 640;

  v8f acc[4];
#pragma unroll
  for (int j = 0; j < 4; ++j)
#pragma unroll
    for (int i = 0; i < 8; ++i) acc[j][i] = 0.f;

  v16h aF[2][4], bF[2];
  {
    int kk = 8 * g;
    bF[0] = frag_from(brow + kk, brow + kk + 16);
#pragma unroll
    for (int mt = 0; mt < 4; ++mt) {
      const half_t* ar = wh + (size_t)(mt * 16 + m) * 640 + kk;
      aF[0][mt] = frag_from(ar, ar + 16);
    }
  }
#pragma unroll
  for (int s = 0; s < 20; ++s) {
    int cur = s & 1, nxt = cur ^ 1;
    if (s + 1 < 20) {                  // prefetch next K-step while computing current
      int kk = 32 * (s + 1) + 8 * g;
      bF[nxt] = frag_from(brow + kk, brow + kk + 16);
#pragma unroll
      for (int mt = 0; mt < 4; ++mt) {
        const half_t* ar = wh + (size_t)(mt * 16 + m) * 640 + kk;
        aF[nxt][mt] = frag_from(ar, ar + 16);
      }
    }
#pragma unroll
    for (int mt = 0; mt < 4; ++mt) acc[mt] = wmma_f16(aF[cur][mt], bF[cur], acc[mt]);
  }
#pragma unroll
  for (int mt = 0; mt < 4; ++mt)
#pragma unroll
    for (int r = 0; r < 8; ++r) {
      int co = mt * 16 + 8 * g + r;
      float sc = bn[co] * rsqrtf(bn[192 + co] + 1e-5f);
      float y = acc[mt][r] * sc + (bn[64 + co] - bn[128 + co] * sc);
      feat[((size_t)img * 112 + col) * 64 + co] = fmaxf(y, 0.f);
    }
}

// -------- stage A: per-pixel L2 normalize over 64ch -> sfT (A rows) / qfh (B cols)
__global__ void k_l2norm(const float* __restrict__ feat, half_t* __restrict__ sfT,
                         half_t* __restrict__ qfh) {
  int idx = blockIdx.x * blockDim.x + threadIdx.x;
  if (idx >= 8000) return;
  int img = idx / 100, p = idx % 100;
  const float* f = feat + ((size_t)img * 112 + p) * 64;
  float ss = 0.f;
  for (int c = 0; c < 64; ++c) ss += f[c] * f[c];
  float inv = 1.f / fmaxf(sqrtf(ss), 1e-8f);
  half_t* dst = (img < 5) ? (sfT + ((size_t)img * 100 + p) * 64)
                          : (qfh + ((size_t)(img - 5) * 112 + p) * 64);
  for (int c = 0; c < 64; ++c) dst[c] = (half_t)(f[c] * inv);
}

// -------- stage A: correlation GEMM (500x64)x(64x100)/query, 4 M-tiles per wave task
__global__ void k_corr(const half_t* __restrict__ sfT, const half_t* __restrict__ qfh,
                       float* __restrict__ corr) {
  int q = blockIdx.x;
  int tid = threadIdx.x, wave = tid >> 5, lane = tid & 31;
  int m = lane & 15, g = lane >> 4;
  for (int t = wave; t < 56; t += 8) {       // 7 N-tiles x 8 groups-of-4-M-tiles
    int nt = t % 7, mg = t / 7;
    int col = nt * 16 + m;
    const half_t* brow = qfh + ((size_t)q * 112 + col) * 64;
    v8f acc[4];
#pragma unroll
    for (int j = 0; j < 4; ++j)
#pragma unroll
      for (int i = 0; i < 8; ++i) acc[j][i] = 0.f;
#pragma unroll
    for (int s = 0; s < 2; ++s) {
      int kk = 32 * s + 8 * g;
      v16h b = frag_from(brow + kk, brow + kk + 16);
#pragma unroll
      for (int j = 0; j < 4; ++j) {
        const half_t* ar = sfT + (size_t)((mg * 4 + j) * 16 + m) * 64 + kk;
        acc[j] = wmma_f16(frag_from(ar, ar + 16), b, acc[j]);
      }
    }
    if (col < 100) {
#pragma unroll
      for (int j = 0; j < 4; ++j)
#pragma unroll
        for (int r = 0; r < 8; ++r) {
          int row = (mg * 4 + j) * 16 + 8 * g + r;
          if (row < 500) {
            int n = row / 100, ij = row % 100;
            corr[(((size_t)q * 5 + n) * 100 + ij) * 100 + col] = acc[j][r];
          }
        }
    }
  }
}

// -------- stage B: transpose (uv)<->(hw) for branch 2
__global__ void k_transpose(const float* __restrict__ src, float* __restrict__ dst) {
  int idx = blockIdx.x * blockDim.x + threadIdx.x;
  if (idx >= 375 * 10000) return;
  int b = idx / 10000, r = idx % 10000;
  dst[(size_t)b * 10000 + (r % 100) * 100 + (r / 100)] = src[idx];
}

// -------- c1 part A: 3x3 conv over first spatial pair (ch=1), bn2, relu
__global__ void k_c1A(const float* __restrict__ x, const float* __restrict__ w9,
                      const float* __restrict__ bn, float* __restrict__ t1) {
  int b = blockIdx.x, tid = threadIdx.x;
  __shared__ float slab[10000];
  for (int i = tid; i < 10000; i += blockDim.x) slab[i] = x[(size_t)b * 10000 + i];
  __syncthreads();
  float sc = bn[0] * rsqrtf(bn[3] + 1e-5f);
  float bi = bn[1] - bn[2] * sc;
  for (int i = tid; i < 10000; i += blockDim.x) {
    int a = i / 100, hb = i % 100, u = a / 10, v = a % 10;
    float acc = 0.f;
#pragma unroll
    for (int du = -1; du <= 1; ++du)
#pragma unroll
      for (int dv = -1; dv <= 1; ++dv) {
        int uu = u + du, vv = v + dv;
        if (uu >= 0 && uu < 10 && vv >= 0 && vv < 10)
          acc += w9[(du + 1) * 3 + (dv + 1)] * slab[(uu * 10 + vv) * 100 + hb];
      }
    t1[(size_t)b * 10000 + i] = fmaxf(acc * sc + bi, 0.f);
  }
}

// -------- c1 part B: 3x3 conv over second pair, bn1, proj->16ch, bnp, relu
// output layout: y16[b][hw][uv][ci]  (ci innermost, ready for uv-conv im2col)
__global__ void k_c1B(const float* __restrict__ t1, const float* __restrict__ w9,
                      const float* __restrict__ bn1, const float* __restrict__ pw,
                      const float* __restrict__ bnp, half_t* __restrict__ y16) {
  int b = blockIdx.x, tid = threadIdx.x;
  __shared__ float slab[10000];
  for (int i = tid; i < 10000; i += blockDim.x) slab[i] = t1[(size_t)b * 10000 + i];
  __syncthreads();
  float s1 = bn1[0] * rsqrtf(bn1[3] + 1e-5f);
  float b1 = bn1[1] - bn1[2] * s1;
  float spn[16], bpn[16];
#pragma unroll
  for (int co = 0; co < 16; ++co) {
    spn[co] = bnp[co] * rsqrtf(bnp[48 + co] + 1e-5f);
    bpn[co] = bnp[16 + co] - bnp[32 + co] * spn[co];
  }
  for (int i = tid; i < 10000; i += blockDim.x) {
    int a = i / 100, hb = i % 100, h = hb / 10, w = hb % 10;
    float acc = 0.f;
#pragma unroll
    for (int dh = -1; dh <= 1; ++dh)
#pragma unroll
      for (int dw = -1; dw <= 1; ++dw) {
        int hh = h + dh, ww = w + dw;
        if (hh >= 0 && hh < 10 && ww >= 0 && ww < 10)
          acc += w9[(dh + 1) * 3 + (dw + 1)] * slab[a * 100 + hh * 10 + ww];
      }
    float t2 = acc * s1 + b1;
    half_t* dst = y16 + (((size_t)b * 100 + hb) * 100 + a) * 16;
#pragma unroll
    for (int co = 0; co < 16; ++co)
      dst[co] = (half_t)fmaxf(spn[co] * (pw[co] * t2) + bpn[co], 0.f);
  }
}

// -------- c2 conv over uv (16->16ch 3x3) as implicit-GEMM WMMA. In: Y[b][hw][uv][ci],
// Out (bn2+relu): Z[b][uv][hw][co].  Weight A-frags hoisted out of the N-tile loop.
__global__ void k_conv2_wmma(const half_t* __restrict__ Y, const half_t* __restrict__ Wp,
                             const float* __restrict__ bn, half_t* __restrict__ Z) {
  int gw = (blockIdx.x * blockDim.x + threadIdx.x) >> 5;
  if (gw >= 37500) return;
  int lane = threadIdx.x & 31, m = lane & 15, g = lane >> 4;
  int b = gw / 100, hw = gw % 100;
  float sc[8], bi[8];
#pragma unroll
  for (int r = 0; r < 8; ++r) {
    int co = 8 * g + r;
    sc[r] = bn[co] * rsqrtf(bn[48 + co] + 1e-5f);
    bi[r] = bn[16 + co] - bn[32 + co] * sc[r];
  }
  const half_t* wrow = Wp + m * 160;
  v16h aF[5];
#pragma unroll
  for (int s = 0; s < 5; ++s) {
    int k0 = 32 * s + 8 * g;
    aF[s] = frag_from(wrow + k0, wrow + k0 + 16);
  }
  const half_t* base = Y + ((size_t)b * 100 + hw) * 100 * 16;
  for (int nt = 0; nt < 7; ++nt) {
    int col = nt * 16 + m;
    bool valid = col < 100;
    int u = col / 10, v = col % 10;
    v8f c;
#pragma unroll
    for (int i = 0; i < 8; ++i) c[i] = 0.f;
#pragma unroll
    for (int s = 0; s < 5; ++s) {
      v16h bf;
#pragma unroll
      for (int i = 0; i < 16; ++i) bf[i] = (_Float16)0;
      if (valid) {
        int tap1 = 2 * s, tap2 = 2 * s + 1;
        {
          int du = tap1 / 3 - 1, dv = tap1 % 3 - 1, uu = u + du, vv = v + dv;
          if (uu >= 0 && uu < 10 && vv >= 0 && vv < 10) {
            v8h lo = *(const v8h*)(base + (uu * 10 + vv) * 16 + 8 * g);
#pragma unroll
            for (int i = 0; i < 8; ++i) bf[i] = lo[i];
          }
        }
        if (tap2 < 9) {
          int du = tap2 / 3 - 1, dv = tap2 % 3 - 1, uu = u + du, vv = v + dv;
          if (uu >= 0 && uu < 10 && vv >= 0 && vv < 10) {
            v8h hi = *(const v8h*)(base + (uu * 10 + vv) * 16 + 8 * g);
#pragma unroll
            for (int i = 0; i < 8; ++i) bf[8 + i] = hi[i];
          }
        }
      }
      c = wmma_f16(aF[s], bf, c);
    }
    if (valid) {
      v8h outv;
#pragma unroll
      for (int r = 0; r < 8; ++r) outv[r] = (half_t)fmaxf(c[r] * sc[r] + bi[r], 0.f);
      *(v8h*)(Z + (((size_t)b * 100 + col) * 100 + hw) * 16 + 8 * g) = outv;
    }
  }
}

// -------- c2 conv over hw via WMMA, then fused bn1 + 16->1 proj + bnp, accumulate.
__global__ void k_conv1_wmma(const half_t* __restrict__ Z, const half_t* __restrict__ Wp,
                             const float* __restrict__ bn1, const float* __restrict__ pw,
                             const float* __restrict__ bnp, float* __restrict__ out,
                             int transposed) {
  int gw = (blockIdx.x * blockDim.x + threadIdx.x) >> 5;
  if (gw >= 37500) return;
  int lane = threadIdx.x & 31, m = lane & 15, g = lane >> 4;
  int b = gw / 100, A = gw % 100;
  float sp = bnp[0] * rsqrtf(bnp[3] + 1e-5f);
  float bp = bnp[1] - bnp[2] * sp;
  float T = 0.f;
#pragma unroll
  for (int co = 0; co < 16; ++co) {
    float s1 = bn1[co] * rsqrtf(bn1[48 + co] + 1e-5f);
    T += pw[co] * (bn1[16 + co] - bn1[32 + co] * s1);
  }
  T = sp * T + bp;
  float coef[8];
#pragma unroll
  for (int r = 0; r < 8; ++r) {
    int co = 8 * g + r;
    coef[r] = sp * pw[co] * (bn1[co] * rsqrtf(bn1[48 + co] + 1e-5f));
  }
  const half_t* wrow = Wp + m * 160;
  v16h aF[5];
#pragma unroll
  for (int s = 0; s < 5; ++s) {
    int k0 = 32 * s + 8 * g;
    aF[s] = frag_from(wrow + k0, wrow + k0 + 16);
  }
  const half_t* base = Z + ((size_t)b * 100 + A) * 100 * 16;
  for (int nt = 0; nt < 7; ++nt) {
    int col = nt * 16 + m;
    bool valid = col < 100;
    int h = col / 10, w = col % 10;
    v8f c;
#pragma unroll
    for (int i = 0; i < 8; ++i) c[i] = 0.f;
#pragma unroll
    for (int s = 0; s < 5; ++s) {
      v16h bf;
#pragma unroll
      for (int i = 0; i < 16; ++i) bf[i] = (_Float16)0;
      if (valid) {
        int tap1 = 2 * s, tap2 = 2 * s + 1;
        {
          int dh = tap1 / 3 - 1, dw = tap1 % 3 - 1, hh = h + dh, ww = w + dw;
          if (hh >= 0 && hh < 10 && ww >= 0 && ww < 10) {
            v8h lo = *(const v8h*)(base + (hh * 10 + ww) * 16 + 8 * g);
#pragma unroll
            for (int i = 0; i < 8; ++i) bf[i] = lo[i];
          }
        }
        if (tap2 < 9) {
          int dh = tap2 / 3 - 1, dw = tap2 % 3 - 1, hh = h + dh, ww = w + dw;
          if (hh >= 0 && hh < 10 && ww >= 0 && ww < 10) {
            v8h hi = *(const v8h*)(base + (hh * 10 + ww) * 16 + 8 * g);
#pragma unroll
            for (int i = 0; i < 8; ++i) bf[8 + i] = hi[i];
          }
        }
      }
      c = wmma_f16(aF[s], bf, c);
    }
    float partial = 0.f;
#pragma unroll
    for (int r = 0; r < 8; ++r) partial += coef[r] * c[r];
    float tot = partial + __shfl_xor(partial, 16, 32);
    if (valid && g == 0) {
      size_t idx = transposed ? (((size_t)b * 100 + col) * 100 + A)
                              : (((size_t)b * 100 + A) * 100 + col);
      out[idx] += tot + T;
    }
  }
}

// -------- stage C: gaussian-normalize + softmax over ij (cols) and kl (rows) + sums
__global__ void k_attn(const float* __restrict__ cca, float* __restrict__ attn_s,
                       float* __restrict__ attn_q) {
  int b = blockIdx.x, tid = threadIdx.x;  // 128 threads
  __shared__ float slab[10000];
  __shared__ float acc[100];
  for (int i = tid; i < 10000; i += 128) slab[i] = cca[(size_t)b * 10000 + i];
  if (tid < 100) acc[tid] = 0.f;
  __syncthreads();
  if (tid < 100) {
    const float* row = slab + tid * 100;
    float s = 0.f, ss = 0.f;
    for (int k = 0; k < 100; ++k) { float x = row[k]; s += x; ss += x * x; }
    float mn = s * 0.01f;
    float var = (ss - 100.f * mn * mn) / 99.f;
    float inv = 1.f / (sqrtf(var + 1e-5f) * 5.f);
    float mx = -1e30f;
    for (int k = 0; k < 100; ++k) mx = fmaxf(mx, (row[k] - mn) * inv);
    float es = 0.f;
    for (int k = 0; k < 100; ++k) es += __expf((row[k] - mn) * inv - mx);
    float ie = 1.f / es;
    for (int k = 0; k < 100; ++k)
      atomicAdd(&acc[k], __expf((row[k] - mn) * inv - mx) * ie);
  }
  __syncthreads();
  if (tid < 100) attn_q[b * 100 + tid] = acc[tid];
  __syncthreads();
  if (tid < 100) acc[tid] = 0.f;
  __syncthreads();
  if (tid < 100) {
    int kl = tid;
    float s = 0.f, ss = 0.f;
    for (int i = 0; i < 100; ++i) { float x = slab[i * 100 + kl]; s += x; ss += x * x; }
    float mn = s * 0.01f;
    float var = (ss - 100.f * mn * mn) / 99.f;
    float inv = 1.f / (sqrtf(var + 1e-5f) * 5.f);
    float mx = -1e30f;
    for (int i = 0; i < 100; ++i) mx = fmaxf(mx, (slab[i * 100 + kl] - mn) * inv);
    float es = 0.f;
    for (int i = 0; i < 100; ++i) es += __expf((slab[i * 100 + kl] - mn) * inv - mx);
    float ie = 1.f / es;
    for (int i = 0; i < 100; ++i)
      atomicAdd(&acc[i], __expf((slab[i * 100 + kl] - mn) * inv - mx) * ie);
  }
  __syncthreads();
  if (tid < 100) attn_s[b * 100 + tid] = acc[tid];
}

// -------- stage C: attention-weighted pooling over pixels
__global__ void k_attpool(const float* __restrict__ attn_s, const float* __restrict__ attn_q,
                          const half_t* __restrict__ xh, float* __restrict__ satt,
                          float* __restrict__ qatt) {
  int idx = blockIdx.x * blockDim.x + threadIdx.x;
  if (idx >= 375 * 640) return;
  int b = idx / 640, c = idx % 640, n = b % 5, q = b / 5;
  float a1 = 0.f, a2 = 0.f;
  for (int p = 0; p < 100; ++p) {
    a1 += attn_s[b * 100 + p] * (float)xh[((size_t)n * 112 + p) * 640 + c];
    a2 += attn_q[b * 100 + p] * (float)xh[((size_t)(5 + q) * 112 + p) * 640 + c];
  }
  satt[idx] = a1 * 0.01f;
  qatt[idx] = a2 * 0.01f;
}

// -------- stage C: cosine similarity / temperature
__global__ void k_sim(const float* __restrict__ satt, const float* __restrict__ qatt,
                      float* __restrict__ out) {
  int b = blockIdx.x, tid = threadIdx.x;
  float num = 0.f, ns = 0.f, nq = 0.f;
  for (int c = tid; c < 640; c += 256) {
    float s = satt[b * 640 + c], qv = qatt[b * 640 + c];
    num += s * qv; ns += s * s; nq += qv * qv;
  }
  __shared__ float sn[256], ss[256], sq[256];
  sn[tid] = num; ss[tid] = ns; sq[tid] = nq;
  __syncthreads();
  for (int o = 128; o > 0; o >>= 1) {
    if (tid < o) { sn[tid] += sn[tid + o]; ss[tid] += ss[tid + o]; sq[tid] += sq[tid + o]; }
    __syncthreads();
  }
  if (tid == 0) {
    float den = fmaxf(sqrtf(ss[0]), 1e-8f) * fmaxf(sqrtf(sq[0]), 1e-8f);
    out[b] = (sn[0] / den) * 5.f;  // /TEMPERATURE(0.2)
  }
}

// -------- qry_pooled = mean over pixels of centered query (exact, from f32 inputs)
__global__ void k_qpool(const float* __restrict__ qry, const float* __restrict__ meanb,
                        float* __restrict__ out) {
  int idx = blockIdx.x * blockDim.x + threadIdx.x;
  if (idx >= 75 * 640) return;
  int q = idx / 640, c = idx % 640;
  float s1 = 0.f, s2 = 0.f;
  for (int p = 0; p < 100; ++p) s1 += qry[((size_t)q * 640 + c) * 100 + p];
  for (int p = 0; p < 100; ++p) s2 += meanb[(5 + q) * 100 + p];
  out[375 + idx] = (s1 - s2) * 0.01f;
}

extern "C" void kernel_launch(void* const* d_in, const int* in_sizes, int n_in,
                              void* d_out, int out_size, void* d_ws, size_t ws_size,
                              hipStream_t stream) {
  (void)in_sizes; (void)n_in; (void)out_size; (void)ws_size;
  const float* spt   = (const float*)d_in[0];
  const float* qry   = (const float*)d_in[1];
  const float* w1x1  = (const float*)d_in[2];
  const float* bn1x1 = (const float*)d_in[3];
  const float* c1w2  = (const float*)d_in[4];
  const float* c1bn2 = (const float*)d_in[5];
  const float* c1w1  = (const float*)d_in[6];
  const float* c1bn1 = (const float*)d_in[7];
  const float* c1pw  = (const float*)d_in[8];
  const float* c1bnp = (const float*)d_in[9];
  const float* c2w2  = (const float*)d_in[10];
  const float* c2bn2 = (const float*)d_in[11];
  const float* c2w1  = (const float*)d_in[12];
  const float* c2bn1 = (const float*)d_in[13];
  const float* c2pw  = (const float*)d_in[14];
  const float* c2bnp = (const float*)d_in[15];
  float* out = (float*)d_out;

  char* base = (char*)d_ws;
  auto carve = [&](size_t bytes) -> char* {
    char* p = base;
    base += (bytes + 255) & ~(size_t)255;
    return p;
  };
  half_t* xh    = (half_t*)carve(sizeof(half_t) * 80 * 112 * 640);
  half_t* wh    = (half_t*)carve(sizeof(half_t) * 64 * 640);
  float*  feat  = (float*) carve(sizeof(float) * 80 * 112 * 64);
  half_t* sfT   = (half_t*)carve(sizeof(half_t) * 512 * 64);
  half_t* qfh   = (half_t*)carve(sizeof(half_t) * 75 * 112 * 64);
  float*  corr  = (float*) carve(sizeof(float) * 375 * 10000);
  float*  xT    = (float*) carve(sizeof(float) * 375 * 10000);
  float*  t1    = (float*) carve(sizeof(float) * 375 * 10000);
  half_t* y16   = (half_t*)carve(sizeof(half_t) * (size_t)375 * 100 * 100 * 16);
  half_t* zbuf  = (half_t*)carve(sizeof(half_t) * (size_t)375 * 100 * 100 * 16);
  float*  cca   = (float*) carve(sizeof(float) * 375 * 10000);
  half_t* Wp2   = (half_t*)carve(sizeof(half_t) * 16 * 160);
  half_t* Wp1   = (half_t*)carve(sizeof(half_t) * 16 * 160);
  float*  meanb = (float*) carve(sizeof(float) * 8000);
  float*  attns = (float*) carve(sizeof(float) * 375 * 100);
  float*  attnq = (float*) carve(sizeof(float) * 375 * 100);
  float*  satt  = (float*) carve(sizeof(float) * 375 * 640);
  float*  qatt  = (float*) carve(sizeof(float) * 375 * 640);

  hipMemsetAsync(xh, 0, sizeof(half_t) * 80 * 112 * 640, stream);
  hipMemsetAsync(sfT, 0, sizeof(half_t) * 512 * 64, stream);
  hipMemsetAsync(qfh, 0, sizeof(half_t) * 75 * 112 * 64, stream);
  hipMemsetAsync(cca, 0, sizeof(float) * 375 * 10000, stream);

  k_pack<<<(46080 + 255) / 256, 256, 0, stream>>>(w1x1, c2w2, c2w1, wh, Wp2, Wp1);
  k_center<<<(8000 + 255) / 256, 256, 0, stream>>>(spt, qry, xh, meanb);
  k_gemm_feat<<<80, 224, 0, stream>>>(wh, xh, bn1x1, feat);
  k_l2norm<<<(8000 + 255) / 256, 256, 0, stream>>>(feat, sfT, qfh);
  k_corr<<<75, 256, 0, stream>>>(sfT, qfh, corr);

  for (int br = 0; br < 2; ++br) {
    const float* xin = corr;
    if (br == 1) {
      k_transpose<<<(3750000 + 255) / 256, 256, 0, stream>>>(corr, xT);
      xin = xT;
    }
    k_c1A<<<375, 256, 0, stream>>>(xin, c1w2, c1bn2, t1);
    k_c1B<<<375, 256, 0, stream>>>(t1, c1w1, c1bn1, c1pw, c1bnp, y16);
    k_conv2_wmma<<<4688, 256, 0, stream>>>(y16, Wp2, c2bn2, zbuf);
    k_conv1_wmma<<<4688, 256, 0, stream>>>(zbuf, Wp1, c2bn1, c2pw, c2bnp, cca, br);
  }

  k_attn<<<375, 128, 0, stream>>>(cca, attns, attnq);
  k_attpool<<<(240000 + 255) / 256, 256, 0, stream>>>(attns, attnq, xh, satt, qatt);
  k_sim<<<375, 256, 0, stream>>>(satt, qatt, out);
  k_qpool<<<(48000 + 255) / 256, 256, 0, stream>>>(qry, meanb, out);
}
